// spiking_transformer_13537736917350
// MI455X (gfx1250) — matrix operand order repaired and checked by reading, not compiled, not run on hardware
//
#include <hip/hip_runtime.h>
#include <hip/hip_bf16.h>

typedef __attribute__((ext_vector_type(16))) _Float16 v16h;
typedef __attribute__((ext_vector_type(8)))  _Float16 v8h;
typedef __attribute__((ext_vector_type(8)))  float    v8f;

#define T_  4
#define B_  4
#define C_  2
#define H_  128
#define W_  128
#define HF_ 128      // E/2 hidden channels
#define E_  256
#define PH_ 130      // padded spike height
#define PW_ 130      // padded spike width
#define EPS_ 1e-5f

// padded LDS tile geometry (halfs): col stride 136 (=272B -> conflict-free b128),
// row stride 18 cols, one t-slice = 3 rows
#define CSTR_ 136
#define RSTR_ (18 * CSTR_)   // 2448
#define TSTR_ (3 * RSTR_)    // 7344

// ---------- fragment load helpers ----------
// A-fragment (global, f16): lane supplies K = {base..+7, base+16..+23}
static __device__ __forceinline__ v16h ld16_gap(const _Float16* p) {
  union { v16h v; v8h h[2]; } u;
  u.h[0] = *(const v8h*)(p);
  u.h[1] = *(const v8h*)(p + 16);
  return u.v;
}
// B-fragment (LDS, f16): lane supplies K = {base..+7, base+16..+23}
static __device__ __forceinline__ v16h ld16_lds(const _Float16* p) {
  union { v16h v; v8h h[2]; } u;
  u.h[0] = *(const v8h*)(p);
  u.h[1] = *(const v8h*)(p + 16);
  return u.v;
}
// async global -> LDS copy of 16B per lane (gfx1250, tracked by ASYNCcnt)
static __device__ __forceinline__ void async_ld16(unsigned lds_off, const void* gaddr) {
  asm volatile("global_load_async_to_lds_b128 %0, %1, off"
               :: "v"(lds_off), "v"(gaddr) : "memory");
}

// ---------- weight prep: fold BN into conv weights ----------
__global__ __launch_bounds__(256) void prep_small(
    const float* __restrict__ wproj, const float* __restrict__ g0,
    const float* __restrict__ b0, const float* __restrict__ m0,
    const float* __restrict__ v0,
    const float* __restrict__ wtemp, const float* __restrict__ gt,
    const float* __restrict__ bt, const float* __restrict__ mt,
    const float* __restrict__ vt,
    float* __restrict__ wpf, float* __restrict__ biasp,
    float* __restrict__ wtf, float* __restrict__ biast)
{
  int i = blockIdx.x * 256 + threadIdx.x;
  if (i < 2304) {                       // w_proj: keep [co][c][tap] layout
    int co = i / 18;
    float sc = g0[co] * rsqrtf(v0[co] + EPS_);
    wpf[i] = wproj[i] * sc;
  } else if (i < 4608) {                // w_temp: [c][ci][tap] -> [c][tap][ci]
    int j = i - 2304;
    int c = j / 1152, r = j - c * 1152;
    int ci = r / 9, tap = r - ci * 9;
    float sc = gt[c] * rsqrtf(vt[c] + EPS_);
    wtf[(c * 9 + tap) * HF_ + ci] = wtemp[(c * HF_ + ci) * 9 + tap] * sc;
  }
  if (i < HF_) biasp[i] = b0[i] - m0[i] * (g0[i] * rsqrtf(v0[i] + EPS_));
  if (i < C_)  biast[i] = bt[i] - mt[i] * (gt[i] * rsqrtf(vt[i] + EPS_));
}

__global__ __launch_bounds__(256) void prep_w1(
    const float* __restrict__ w1, const float* __restrict__ g1,
    const float* __restrict__ b1, const float* __restrict__ m1,
    const float* __restrict__ v1,
    _Float16* __restrict__ w1f, float* __restrict__ bias1)
{
  int i = blockIdx.x * 256 + threadIdx.x;   // 256*9*128 = 294912
  if (i < E_ * 9 * HF_) {
    int co = i / (9 * HF_);
    int r  = i - co * (9 * HF_);
    int tap = r / HF_, ci = r - tap * HF_;
    float sc = g1[co] * rsqrtf(v1[co] + EPS_);
    // src layout [co][ci][3][3]; dst layout [co][tap][ci]
    w1f[i] = (_Float16)(w1[(co * HF_ + ci) * 9 + tap] * sc);
  }
  if (i < E_) bias1[i] = b1[i] - m1[i] * (g1[i] * rsqrtf(v1[i] + EPS_));
}

__global__ __launch_bounds__(256) void prep_wr(
    const float* __restrict__ wres, const float* __restrict__ gr,
    const float* __restrict__ br, const float* __restrict__ mr,
    const float* __restrict__ vr,
    _Float16* __restrict__ wrf, float* __restrict__ biasr)
{
  int i = blockIdx.x * 256 + threadIdx.x;   // 256*128 = 32768
  if (i < E_ * HF_) {
    int co = i / HF_;
    float sc = gr[co] * rsqrtf(vr[co] + EPS_);
    wrf[i] = (_Float16)(wres[i] * sc);
  }
  if (i < E_) biasr[i] = br[i] - mr[i] * (gr[i] * rsqrtf(vr[i] + EPS_));
}

// ---------- recurrent stage 1: conv3x3 (2->128) + BN + LIF -> spikes (f16 NHWC padded) ----------
__global__ __launch_bounds__(256) void conv_proj_lif(
    const float* __restrict__ xt, const float* __restrict__ xout,
    const float* __restrict__ wpf, const float* __restrict__ biasp,
    float* __restrict__ vp, _Float16* __restrict__ spk)
{
  int pix = blockIdx.x * 256 + threadIdx.x;   // B*H*W = 65536
  int b = pix >> 14;
  int y = (pix >> 7) & 127;
  int x = pix & 127;

  float xin[18];
  #pragma unroll
  for (int c = 0; c < C_; ++c)
    #pragma unroll
    for (int dy = 0; dy < 3; ++dy)
      #pragma unroll
      for (int dx = 0; dx < 3; ++dx) {
        int gy = y + dy - 1, gx = x + dx - 1;
        float v = 0.f;
        if (gy >= 0 && gy < H_ && gx >= 0 && gx < W_) {
          int idx = ((b * C_ + c) * H_ + gy) * W_ + gx;
          v = xt[idx] + xout[idx];         // recurrent input injection
        }
        xin[c * 9 + dy * 3 + dx] = v;
      }

  _Float16* sp = spk + (size_t)((b * PH_ + y + 1) * PW_ + x + 1) * HF_;
  for (int co = 0; co < HF_; ++co) {
    float h = biasp[co];
    const float* wr = wpf + co * 18;
    #pragma unroll
    for (int k = 0; k < 18; ++k) h = fmaf(wr[k], xin[k], h);
    int vidx = ((b * HF_ + co) * H_ + y) * W_ + x;
    float v = vp[vidx];
    v += (h - v) * 0.5f;                   // charge, tau = 2
    float s = (v >= 1.0f) ? 1.0f : 0.0f;   // fire
    vp[vidx] = (1.0f - s) * v;             // hard reset
    sp[co] = (_Float16)s;                  // exact 0/1 in f16
  }
}

// ---------- recurrent stage 2: conv3x3 (128->2) + BN + LIF -> x_out feedback ----------
__global__ __launch_bounds__(256) void conv_temp_lif(
    const _Float16* __restrict__ spk, const float* __restrict__ wtf,
    const float* __restrict__ biast, float* __restrict__ vtm,
    float* __restrict__ xout)
{
  int pix = blockIdx.x * 256 + threadIdx.x;
  int b = pix >> 14;
  int y = (pix >> 7) & 127;
  int x = pix & 127;

  float a0 = biast[0], a1 = biast[1];
  const _Float16* base = spk + (size_t)((b * PH_ + y) * PW_ + x) * HF_;
  for (int tap = 0; tap < 9; ++tap) {
    int dy = tap / 3, dx = tap - dy * 3;
    const _Float16* s = base + (size_t)(dy * PW_ + dx) * HF_;
    const float* w0 = wtf + tap * HF_;
    const float* w1 = wtf + (9 + tap) * HF_;
    for (int ci = 0; ci < HF_; ci += 8) {
      v8h sv = *(const v8h*)(s + ci);
      #pragma unroll
      for (int j = 0; j < 8; ++j) {
        float f = (float)sv[j];
        a0 = fmaf(w0[ci + j], f, a0);
        a1 = fmaf(w1[ci + j], f, a1);
      }
    }
  }
  #pragma unroll
  for (int c = 0; c < C_; ++c) {
    float h = c ? a1 : a0;
    int idx = ((b * C_ + c) * H_ + y) * W_ + x;
    float v = vtm[idx];
    v += (h - v) * 0.5f;
    float s = (v >= 1.0f) ? 1.0f : 0.0f;
    vtm[idx] = (1.0f - s) * v;
    xout[idx] = s;
  }
}

// ---------- main WMMA kernel ----------
// Block: 256 threads (8 waves). All 4 timesteps' spike tiles staged in LDS via
// async-to-LDS; t is the innermost WMMA loop so each A-fragment load feeds
// 8 WMMAs (4 t x 2 co-tiles). Residual 1x1 conv + LIF + output in epilogue.
__global__ __launch_bounds__(256) void conv_main_res_lif(
    const _Float16* __restrict__ spikes, const _Float16* __restrict__ w1f,
    const float* __restrict__ bias1, const _Float16* __restrict__ wrf,
    const float* __restrict__ biasr, float* __restrict__ out)
{
  __shared__ __align__(16) _Float16 tile[4 * TSTR_];   // 58752 B, [t][row][col(pad 136)][ch]

  const int tid   = threadIdx.x;
  const int wave  = tid >> 5;
  const int lane  = tid & 31;
  const int lhalf = lane >> 4;     // which 16-lane half
  const int lidx  = lane & 15;
  const int bx = blockIdx.x;       // 4096 = B * H * (W/16)
  const int x0 = (bx & 7) * 16;
  const int y  = (bx >> 3) & 127;
  const int b  = bx >> 10;
  const int co0 = wave * 32;

  // ---- async fill: 4 x (3 rows x 2304 halfs) global -> LDS ----
  const unsigned ldsBase = (unsigned)(size_t)(&tile[0]);
  #pragma unroll 1
  for (int t = 0; t < T_; ++t) {
    const _Float16* srcT = spikes + (size_t)((t * B_ + b) * PH_ + y) * (PW_ * HF_)
                                  + (size_t)x0 * HF_;
    for (int v = tid; v < 864; v += 256) {       // 864 x 16B vectors per t
      int row = v / 288, rem = v - row * 288;    // 288 = 18 cols * 16 vec/col
      int col = rem >> 4, cv = rem & 15;
      const _Float16* g = srcT + (size_t)row * (PW_ * HF_) + rem * 8;
      unsigned lo = ldsBase +
          (unsigned)(((t * 3 + row) * RSTR_ + col * CSTR_ + cv * 8) * 2);
      async_ld16(lo, (const void*)g);
    }
  }

  // bias preload while the async DMA runs
  float b1v[2][8], brv[2][8];
  #pragma unroll
  for (int ct = 0; ct < 2; ++ct)
    #pragma unroll
    for (int r = 0; r < 8; ++r) {
      int co = co0 + ct * 16 + r + lhalf * 8;   // C/D layout: vgpr r, lanes16-31 -> M=r+8
      b1v[ct][r] = bias1[co];
      brv[ct][r] = biasr[co];
    }

  asm volatile("s_wait_asynccnt 0x0" ::: "memory");
  __syncthreads();

  // ---- main conv3x3 K-accumulation, t innermost ----
  v8f z = {0.f,0.f,0.f,0.f,0.f,0.f,0.f,0.f};
  v8f accA[T_][2];
  #pragma unroll
  for (int t = 0; t < T_; ++t) { accA[t][0] = z; accA[t][1] = z; }

  #pragma unroll 1
  for (int dy = 0; dy < 3; ++dy) {
    #pragma unroll 1
    for (int dx = 0; dx < 3; ++dx) {
      const int tap = dy * 3 + dx;
      const _Float16* a0base = w1f + (size_t)((co0 + lidx) * 9 + tap) * HF_ + lhalf * 8;
      const _Float16* a1base = a0base + (size_t)16 * 9 * HF_;
      const int bcol = (dx + lidx) * CSTR_ + lhalf * 8;
      #pragma unroll
      for (int kc = 0; kc < 4; ++kc) {
        const int c0 = kc * 32;
        v16h a0 = ld16_gap(a0base + c0);
        v16h a1 = ld16_gap(a1base + c0);
        #pragma unroll
        for (int t = 0; t < T_; ++t) {
          v16h bb = ld16_lds(&tile[(t * 3 + dy) * RSTR_ + bcol + c0]);
          accA[t][0] = __builtin_amdgcn_wmma_f32_16x16x32_f16(false, a0, false, bb,
                                                              (short)0, accA[t][0], false, false);
          accA[t][1] = __builtin_amdgcn_wmma_f32_16x16x32_f16(false, a1, false, bb,
                                                              (short)0, accA[t][1], false, false);
        }
      }
    }
  }

  // ---- epilogue per t: residual 1x1 WMMA, bias + dual LIF, store s_a + s_r ----
  float vaS[2][8], vrS[2][8];
  #pragma unroll
  for (int ct = 0; ct < 2; ++ct)
    #pragma unroll
    for (int r = 0; r < 8; ++r) { vaS[ct][r] = 0.f; vrS[ct][r] = 0.f; }

  const int ccol = (1 + lidx) * CSTR_ + lhalf * 8;   // center tap (dy=1,dx=1)
  #pragma unroll 1
  for (int t = 0; t < T_; ++t) {
    v8f accR0 = z, accR1 = z;
    #pragma unroll
    for (int kc = 0; kc < 4; ++kc) {
      const int c0 = kc * 32;
      v16h r0 = ld16_gap(wrf + (size_t)(co0 + lidx) * HF_ + lhalf * 8 + c0);
      v16h r1 = ld16_gap(wrf + (size_t)(co0 + 16 + lidx) * HF_ + lhalf * 8 + c0);
      v16h bb = ld16_lds(&tile[(t * 3 + 1) * RSTR_ + ccol + c0]);
      accR0 = __builtin_amdgcn_wmma_f32_16x16x32_f16(false, r0, false, bb,
                                                     (short)0, accR0, false, false);
      accR1 = __builtin_amdgcn_wmma_f32_16x16x32_f16(false, r1, false, bb,
                                                     (short)0, accR1, false, false);
    }
    #pragma unroll
    for (int ct = 0; ct < 2; ++ct) {
      v8f aA = accA[t][ct];
      v8f aR = ct ? accR1 : accR0;
      #pragma unroll
      for (int r = 0; r < 8; ++r) {
        int co = co0 + ct * 16 + r + lhalf * 8;
        float ha = aA[r] + b1v[ct][r];
        float va = vaS[ct][r]; va += (ha - va) * 0.5f;
        float sa = (va >= 1.0f) ? 1.0f : 0.0f;
        vaS[ct][r] = (1.0f - sa) * va;
        float hr = aR[r] + brv[ct][r];
        float vv = vrS[ct][r]; vv += (hr - vv) * 0.5f;
        float sr = (vv >= 1.0f) ? 1.0f : 0.0f;
        vrS[ct][r] = (1.0f - sr) * vv;
        out[(size_t)((t * B_ + b) * E_ + co) * (H_ * W_) + y * W_ + x0 + lidx] = sa + sr;
      }
    }
  }
}

// ---------- host launch ----------
extern "C" void kernel_launch(void* const* d_in, const int* in_sizes, int n_in,
                              void* d_out, int out_size, void* d_ws, size_t ws_size,
                              hipStream_t stream) {
  (void)in_sizes; (void)n_in; (void)out_size; (void)ws_size;

  const float* x     = (const float*)d_in[0];
  const float* wproj = (const float*)d_in[1];
  const float* g0 = (const float*)d_in[2];
  const float* b0 = (const float*)d_in[3];
  const float* m0 = (const float*)d_in[4];
  const float* v0 = (const float*)d_in[5];
  const float* wtemp = (const float*)d_in[6];
  const float* gt = (const float*)d_in[7];
  const float* bt = (const float*)d_in[8];
  const float* mt = (const float*)d_in[9];
  const float* vt = (const float*)d_in[10];
  const float* w1in = (const float*)d_in[11];
  const float* g1 = (const float*)d_in[12];
  const float* b1 = (const float*)d_in[13];
  const float* m1 = (const float*)d_in[14];
  const float* v1 = (const float*)d_in[15];
  const float* wres = (const float*)d_in[16];
  const float* gr = (const float*)d_in[17];
  const float* br = (const float*)d_in[18];
  const float* mr = (const float*)d_in[19];
  const float* vr = (const float*)d_in[20];

  char* base = (char*)d_ws;
  size_t off = 0;
  auto carve = [&](size_t bytes) -> char* {
    char* p = base + off;
    off = (off + bytes + 255) & ~(size_t)255;
    return p;
  };

  const size_t spkT   = (size_t)B_ * PH_ * PW_ * HF_;          // elems per t
  const size_t spkAll = (size_t)T_ * spkT * sizeof(_Float16);  // 69.2 MB
  _Float16* spikes = (_Float16*)carve(spkAll);
  float* vp   = (float*)carve((size_t)B_ * HF_ * H_ * W_ * 4); // 33.5 MB
  float* vtm  = (float*)carve((size_t)B_ * C_ * H_ * W_ * 4);
  float* xout = (float*)carve((size_t)B_ * C_ * H_ * W_ * 4);
  float* wpf    = (float*)carve(2304 * 4);
  float* biasp  = (float*)carve(512);
  float* wtf    = (float*)carve(2304 * 4);
  float* biast  = (float*)carve(512);
  _Float16* w1f = (_Float16*)carve((size_t)E_ * 9 * HF_ * 2);
  float* bias1  = (float*)carve(E_ * 4);
  _Float16* wrf = (_Float16*)carve((size_t)E_ * HF_ * 2);
  float* biasr  = (float*)carve(E_ * 4);

  // zero spike halo + LIF/recurrent state (graph-capturable)
  hipMemsetAsync(spikes, 0, spkAll, stream);
  hipMemsetAsync(vp, 0, (size_t)B_ * HF_ * H_ * W_ * 4, stream);
  hipMemsetAsync(vtm, 0, (size_t)B_ * C_ * H_ * W_ * 4, stream);
  hipMemsetAsync(xout, 0, (size_t)B_ * C_ * H_ * W_ * 4, stream);

  prep_small<<<18, 256, 0, stream>>>(wproj, g0, b0, m0, v0, wtemp, gt, bt, mt, vt,
                                     wpf, biasp, wtf, biast);
  prep_w1<<<1152, 256, 0, stream>>>(w1in, g1, b1, m1, v1, w1f, bias1);
  prep_wr<<<128, 256, 0, stream>>>(wres, gr, br, mr, vr, wrf, biasr);

  const size_t xT = (size_t)B_ * C_ * H_ * W_;   // 131072
  for (int t = 0; t < T_; ++t) {
    conv_proj_lif<<<256, 256, 0, stream>>>(x + (size_t)t * xT, xout, wpf, biasp,
                                           vp, spikes + (size_t)t * spkT);
    conv_temp_lif<<<256, 256, 0, stream>>>(spikes + (size_t)t * spkT, wtf, biast,
                                           vtm, xout);
  }

  conv_main_res_lif<<<4096, 256, 0, stream>>>(spikes, w1f, bias1, wrf, biasr,
                                              (float*)d_out);
}
